// EfficientSelfAttention_45947560133074
// MI455X (gfx1250) — matrix operand order repaired and checked by reading, not compiled, not run on hardware
//
#include <hip/hip_runtime.h>
#include <hip/hip_bf16.h>

typedef __bf16 bf16_t;
typedef __attribute__((ext_vector_type(8)))  __bf16 v8bf;
typedef __attribute__((ext_vector_type(16))) __bf16 v16bf;
typedef __attribute__((ext_vector_type(8)))  float  v8f;
typedef unsigned int u32x4 __attribute__((ext_vector_type(4)));
typedef int          i32x4 __attribute__((ext_vector_type(4)));
typedef int          i32x8 __attribute__((ext_vector_type(8)));

#define DIM 256
#define HEAD 2
#define BB 8
#define NN 2048
#define QKVW 512            // HEAD*DIM
#define TOKENS (BB*NN)      // 16384
#define JT 32               // attention j-block

// ---------------- WMMA fragment helpers (gfx1250 wave32 layouts) -------------

// A fragment 16x32 bf16 from row-major [M][K] buffer (ld elements)
__device__ __forceinline__ v16bf load_a_frag(const bf16_t* p, int ld) {
  int lane = threadIdx.x & 31;
  int m  = lane & 15;
  int lh = lane >> 4;
  const bf16_t* r = p + m * ld + lh * 8;
  v8bf lo = *(const v8bf*)(r);        // K = 8*lh + 0..7
  v8bf hi = *(const v8bf*)(r + 16);   // K = 16 + 8*lh + 0..7
  v16bf a;
#pragma unroll
  for (int i = 0; i < 8; ++i) { a[i] = lo[i]; a[8 + i] = hi[i]; }
  return a;
}

// B fragment 32x16 bf16 from N-major [N][K] buffer (ld elements)
__device__ __forceinline__ v16bf load_b_frag(const bf16_t* p, int ld) {
  int lane = threadIdx.x & 31;
  int n  = lane & 15;
  int lh = lane >> 4;
  const bf16_t* r = p + (size_t)n * ld + lh * 8;
  v8bf lo = *(const v8bf*)(r);        // K = 8*lh + 0..7
  v8bf hi = *(const v8bf*)(r + 16);   // K = 16 + 8*lh + 0..7
  v16bf b;
#pragma unroll
  for (int i = 0; i < 8; ++i) { b[i] = lo[i]; b[8 + i] = hi[i]; }
  return b;
}

__device__ __forceinline__ v8f wmma_bf16(v16bf a, v16bf b, v8f c) {
  return __builtin_amdgcn_wmma_f32_16x16x32_bf16(false, a, false, b, (short)0, c, false, false);
}

// ---------------- Tensor Data Mover: 2D tile Global -> LDS -------------------
// Builds D# per CDNA5 ISA ch.8 and issues TENSOR_LOAD_TO_LDS.
// dims/tiles in elements (data_size = 2 bytes), stride0 in elements.
__device__ __forceinline__ void tdm_load_2d(unsigned lds_off, unsigned long long gaddr,
                                            unsigned dim0, unsigned dim1,
                                            unsigned tile0, unsigned tile1,
                                            unsigned long long stride0) {
  u32x4 g0;
  g0[0] = 1u;                                             // count=1, user mode
  g0[1] = lds_off;                                        // lds_addr (bytes)
  g0[2] = (unsigned)(gaddr & 0xFFFFFFFFull);              // global_addr[31:0]
  g0[3] = (unsigned)((gaddr >> 32) & 0x1FFFFFFull)        // global_addr[56:32]
          | (2u << 30);                                   // type = 2 ("image")
  i32x8 g1;
  g1[0] = (int)(1u << 16);                                // data_size=1 (2 bytes)
  g1[1] = (int)((dim0 & 0xFFFFu) << 16);                  // tensor_dim0[15:0]
  g1[2] = (int)(((dim0 >> 16) & 0xFFFFu) | ((dim1 & 0xFFFFu) << 16));
  g1[3] = (int)(((dim1 >> 16) & 0xFFFFu) | ((tile0 & 0xFFFFu) << 16));
  g1[4] = (int)(tile1 & 0xFFFFu);                         // tile_dim1 (tile_dim2=0)
  g1[5] = (int)(unsigned)(stride0 & 0xFFFFFFFFull);       // tensor_dim0_stride[31:0]
  g1[6] = (int)(unsigned)((stride0 >> 32) & 0xFFFFull);   // stride[47:32]
  g1[7] = 0;
  i32x4 gz = {0, 0, 0, 0};
#if __clang_major__ >= 23
  i32x8 gz8 = {0, 0, 0, 0, 0, 0, 0, 0};
  __builtin_amdgcn_tensor_load_to_lds(g0, g1, gz, gz, gz8, 0);
#else
  __builtin_amdgcn_tensor_load_to_lds(g0, g1, gz, gz, 0);
#endif
}

__device__ __forceinline__ unsigned lds_offset_of(const void* p) {
  // On gfx1250 the low 32 bits of a generic LDS address are the LDS byte offset.
  return (unsigned)(unsigned long long)(uintptr_t)p;
}

// ---------------- Stage 1: conversions / weight packing ----------------------

__global__ void cvt_x_kernel(const float* __restrict__ x, bf16_t* __restrict__ xb, int n) {
  for (int i = blockIdx.x * blockDim.x + threadIdx.x; i < n; i += gridDim.x * blockDim.x)
    xb[i] = (bf16_t)x[i];
}

// Wqkvt: [1536][256] N-major (rows 0..511 = Wq cols, 512..1023 = Wk cols, 1024..1535 = Wv cols)
// Wmt:   [256][512]  N-major (Wmt[n][k] = Wm[k][n])
__global__ void pack_w_kernel(const float* __restrict__ Wq, const float* __restrict__ Wk,
                              const float* __restrict__ Wv, const float* __restrict__ Wm,
                              bf16_t* __restrict__ Wqkvt, bf16_t* __restrict__ Wmt) {
  const int NQKV = 3 * QKVW * DIM;       // 393216
  const int NM   = DIM * QKVW;           // 131072
  for (int i = blockIdx.x * blockDim.x + threadIdx.x; i < NQKV + NM;
       i += gridDim.x * blockDim.x) {
    if (i < NQKV) {
      int nrow = i >> 8;                 // 0..1535
      int k    = i & 255;
      int w = nrow >> 9;
      int c = nrow & 511;
      const float* W = (w == 0) ? Wq : (w == 1 ? Wk : Wv);
      Wqkvt[i] = (bf16_t)W[k * QKVW + c];
    } else {
      int j = i - NQKV;
      int nrow = j >> 9;                 // 0..255
      int k    = j & 511;
      Wmt[j] = (bf16_t)Wm[k * DIM + nrow];
    }
  }
}

// ---------------- Stage 2: QKV projection GEMM -------------------------------
__global__ void qkv_gemm_kernel(const bf16_t* __restrict__ xb, const bf16_t* __restrict__ Wqkvt,
                                const float* __restrict__ bq, const float* __restrict__ bk,
                                const float* __restrict__ bv,
                                bf16_t* __restrict__ Qb, bf16_t* __restrict__ Kb,
                                bf16_t* __restrict__ Vt) {
  int widx = threadIdx.x >> 5;
  int tn = blockIdx.x;                   // 0..95   (1536/16)
  int tm = blockIdx.y * 4 + widx;        // 0..1023 (16384/16)
  int lane = threadIdx.x & 31;
  int n  = lane & 15;
  int lh = lane >> 4;

  const bf16_t* ap = xb    + (size_t)(tm * 16) * DIM;
  const bf16_t* bp = Wqkvt + (size_t)(tn * 16) * DIM;
  v8f acc = {};
#pragma unroll
  for (int k0 = 0; k0 < DIM; k0 += 32)
    acc = wmma_bf16(load_a_frag(ap + k0, DIM), load_b_frag(bp + k0, DIM), acc);

  int col = tn * 16 + n;
  float bias = (col < QKVW) ? bq[col] : (col < 2 * QKVW ? bk[col - QKVW] : bv[col - 2 * QKVW]);
#pragma unroll
  for (int r = 0; r < 8; ++r) {
    int row = tm * 16 + 8 * lh + r;      // global token
    bf16_t val = (bf16_t)(acc[r] + bias);
    if (col < QKVW) {
      Qb[(size_t)row * QKVW + col] = val;
    } else if (col < 2 * QKVW) {
      Kb[(size_t)row * QKVW + (col - QKVW)] = val;
    } else {
      int c  = col - 2 * QKVW;
      int hh = c >> 8, d = c & 255;
      int bt = row >> 11, tok = row & 2047;
      Vt[((size_t)((bt * HEAD + hh) << 8) + d) * NN + tok] = val;
    }
  }
}

// ---------------- Stage 3: flash attention with TDM-staged K/V ---------------
// 8 waves/block share (b,h); each wave owns 16 query rows. K/V tiles for the
// shared j-stream are DMA'd into LDS by the Tensor Data Mover, double-buffered.
__global__ __launch_bounds__(256) void attn_kernel(
    const bf16_t* __restrict__ Qb, const bf16_t* __restrict__ Kb,
    const bf16_t* __restrict__ Vt, const unsigned char* __restrict__ mask,
    bf16_t* __restrict__ Ob) {
  __shared__ __align__(16) bf16_t Ktile[2][JT * DIM];   // [32 tok][256 d]  16KB x2
  __shared__ __align__(16) bf16_t Vtile[2][DIM * JT];   // [256 d][32 tok]  16KB x2
  __shared__ __align__(16) bf16_t p_all[8][16 * JT];    // per-wave P tiles  8KB

  int widx = threadIdx.x >> 5;
  bf16_t* p_lds = p_all[widx];

  int wid = blockIdx.x * 8 + widx;       // 0..2047
  int qb  = wid & 127;                   // query block (2048/16)
  int bh  = wid >> 7;                    // 0..15, uniform across the block
  int bbatch = bh >> 1, hh = bh & 1;
  int lane = threadIdx.x & 31;
  int n  = lane & 15;
  int lh = lane >> 4;

  const unsigned long long kgbase =
      (unsigned long long)(uintptr_t)(Kb + (size_t)(bbatch * NN) * QKVW + hh * DIM);
  const unsigned long long vgbase =
      (unsigned long long)(uintptr_t)(Vt + (size_t)((bbatch * HEAD + hh) << 8) * NN);

  const bf16_t* qptr = Qb + (size_t)(bbatch * NN + qb * 16) * QKVW + hh * DIM;
  v16bf aq[8];
#pragma unroll
  for (int kk = 0; kk < 8; ++kk) aq[kk] = load_a_frag(qptr + kk * 32, QKVW);

  v8f o_acc[16];
#pragma unroll
  for (int t = 0; t < 16; ++t) o_acc[t] = (v8f){};
  float m_run[8], l_run[8];
#pragma unroll
  for (int r = 0; r < 8; ++r) { m_run[r] = -3.0e38f; l_run[r] = 0.f; }

  const float scale = 0.0625f;           // DIM^-0.5 = 1/16
  const unsigned char* mrow = mask + bbatch * NN;

  // Prologue: DMA first K/V tiles into buffer 0 (wave 0 only; TDM is per-wave).
  if (widx == 0) {
    tdm_load_2d(lds_offset_of(Ktile[0]), kgbase, DIM, JT, DIM, JT, QKVW);
    tdm_load_2d(lds_offset_of(Vtile[0]), vgbase, JT, DIM, JT, DIM, NN);
  }

  for (int j0 = 0; j0 < NN; j0 += JT) {
    int cur = (j0 >> 5) & 1;
    if (widx == 0) {
      if (j0 + JT < NN) {
        tdm_load_2d(lds_offset_of(Ktile[1 - cur]), kgbase + (size_t)(j0 + JT) * QKVW * 2,
                    DIM, JT, DIM, JT, QKVW);
        tdm_load_2d(lds_offset_of(Vtile[1 - cur]), vgbase + (size_t)(j0 + JT) * 2,
                    JT, DIM, JT, DIM, NN);
        __builtin_amdgcn_s_wait_tensorcnt(2);   // current pair complete, next in flight
      } else {
        __builtin_amdgcn_s_wait_tensorcnt(0);
      }
    }
    __syncthreads();                            // publish buffer `cur` to all waves

    const bf16_t* kt = Ktile[cur];
    const bf16_t* vt = Vtile[cur];

    float sv[2][8];
#pragma unroll
    for (int jt = 0; jt < 2; ++jt) {
      v8f s = {};
#pragma unroll
      for (int kk = 0; kk < 8; ++kk)
        s = wmma_bf16(aq[kk], load_b_frag(kt + (size_t)(jt * 16) * DIM + kk * 32, DIM), s);
      int j = j0 + jt * 16 + n;
      bool msk = mrow[j] != 0;
#pragma unroll
      for (int r = 0; r < 8; ++r) sv[jt][r] = msk ? -32768.f : s[r] * scale;
    }
    // online softmax: rows (8*lh + r) live across the 16 lanes of each half
    float alpha[8];
#pragma unroll
    for (int r = 0; r < 8; ++r) {
      float mc = fmaxf(sv[0][r], sv[1][r]);
#pragma unroll
      for (int off = 1; off < 16; off <<= 1) mc = fmaxf(mc, __shfl_xor(mc, off, 32));
      float mn = fmaxf(m_run[r], mc);
      alpha[r] = __expf(m_run[r] - mn);
      m_run[r] = mn;
      float p0 = __expf(sv[0][r] - mn);
      float p1 = __expf(sv[1][r] - mn);
      float rs = p0 + p1;
#pragma unroll
      for (int off = 1; off < 16; off <<= 1) rs += __shfl_xor(rs, off, 32);
      l_run[r] = l_run[r] * alpha[r] + rs;
      p_lds[(8 * lh + r) * JT + n]      = (bf16_t)p0;   // P tile row-major [16][32]
      p_lds[(8 * lh + r) * JT + 16 + n] = (bf16_t)p1;
    }
    __asm__ volatile("s_wait_dscnt 0" ::: "memory");    // intra-wave LDS RAW
#pragma unroll
    for (int t = 0; t < 16; ++t) {
#pragma unroll
      for (int r = 0; r < 8; ++r) o_acc[t][r] *= alpha[r];
    }
    v16bf ap = load_a_frag(p_lds, JT);
#pragma unroll
    for (int t = 0; t < 16; ++t)
      o_acc[t] = wmma_bf16(ap, load_b_frag(vt + (size_t)(t * 16) * JT, JT), o_acc[t]);

    __syncthreads();                            // all waves done reading `cur`
  }

  float rl[8];
#pragma unroll
  for (int r = 0; r < 8; ++r) rl[r] = 1.f / l_run[r];
#pragma unroll
  for (int t = 0; t < 16; ++t) {
#pragma unroll
    for (int r = 0; r < 8; ++r) {
      int row = bbatch * NN + qb * 16 + 8 * lh + r;
      Ob[(size_t)row * QKVW + hh * DIM + t * 16 + n] = (bf16_t)(o_acc[t][r] * rl[r]);
    }
  }
}

// ---------------- Stage 4: output projection ---------------------------------
__global__ void out_gemm_kernel(const bf16_t* __restrict__ Ob, const bf16_t* __restrict__ Wmt,
                                const float* __restrict__ bm, float* __restrict__ out) {
  int widx = threadIdx.x >> 5;
  int tn = blockIdx.x;                   // 0..15   (256/16)
  int tm = blockIdx.y * 4 + widx;        // 0..1023
  int lane = threadIdx.x & 31;
  int n  = lane & 15;
  int lh = lane >> 4;

  const bf16_t* ap = Ob  + (size_t)(tm * 16) * QKVW;
  const bf16_t* bp = Wmt + (size_t)(tn * 16) * QKVW;
  v8f acc = {};
#pragma unroll
  for (int k0 = 0; k0 < QKVW; k0 += 32)
    acc = wmma_bf16(load_a_frag(ap + k0, QKVW), load_b_frag(bp + k0, QKVW), acc);

  int col = tn * 16 + n;
  float bias = bm[col];
#pragma unroll
  for (int r = 0; r < 8; ++r) {
    int row = tm * 16 + 8 * lh + r;
    out[(size_t)row * DIM + col] = acc[r] + bias;
  }
}

// ---------------- Stage 5: attn_mask broadcast -------------------------------
__global__ void mask_out_kernel(const unsigned char* __restrict__ mask, float* __restrict__ om, int n) {
  for (int i = blockIdx.x * blockDim.x + threadIdx.x; i < n; i += gridDim.x * blockDim.x)
    om[i] = mask[i >> 8] ? 1.f : 0.f;
}

// ---------------- launch -----------------------------------------------------

extern "C" void kernel_launch(void* const* d_in, const int* in_sizes, int n_in,
                              void* d_out, int out_size, void* d_ws, size_t ws_size,
                              hipStream_t stream) {
  const float* x    = (const float*)d_in[0];
  const unsigned char* mask = (const unsigned char*)d_in[1];
  const float* Wq = (const float*)d_in[2];
  const float* bq = (const float*)d_in[3];
  const float* Wk = (const float*)d_in[4];
  const float* bk = (const float*)d_in[5];
  const float* Wv = (const float*)d_in[6];
  const float* bv = (const float*)d_in[7];
  const float* Wm = (const float*)d_in[8];
  const float* bm = (const float*)d_in[9];
  // d_in[10] = pool (== 1): pooling is identity for this config.

  char* ws = (char*)d_ws;
  bf16_t* xb    = (bf16_t*)(ws);                              //  8 MB
  bf16_t* Wqkvt = (bf16_t*)(ws + (size_t)(8  << 20));         //  768 KB
  bf16_t* Wmt   = (bf16_t*)(ws + (size_t)(9  << 20));         //  256 KB
  bf16_t* Qb    = (bf16_t*)(ws + (size_t)(10 << 20));         // 16 MB
  bf16_t* Kb    = (bf16_t*)(ws + (size_t)(26 << 20));         // 16 MB
  bf16_t* Vt    = (bf16_t*)(ws + (size_t)(42 << 20));         // 16 MB
  bf16_t* Ob    = (bf16_t*)(ws + (size_t)(58 << 20));         // 16 MB

  float* out   = (float*)d_out;
  float* out_m = out + (size_t)TOKENS * DIM;

  cvt_x_kernel<<<4096, 256, 0, stream>>>(x, xb, TOKENS * DIM);
  pack_w_kernel<<<2048, 256, 0, stream>>>(Wq, Wk, Wv, Wm, Wqkvt, Wmt);
  qkv_gemm_kernel<<<dim3(96, 256), 128, 0, stream>>>(xb, Wqkvt, bq, bk, bv, Qb, Kb, Vt);
  attn_kernel<<<256, 256, 0, stream>>>(Qb, Kb, Vt, mask, Ob);
  out_gemm_kernel<<<dim3(16, 256), 128, 0, stream>>>(Ob, Wmt, bm, out);
  mask_out_kernel<<<4096, 256, 0, stream>>>(mask, out_m, TOKENS * DIM);
}
